// RepLKSE2D_12987981103619
// MI455X (gfx1250) — matrix operand order repaired and checked by reading, not compiled, not run on hardware
//
#include <hip/hip_runtime.h>
#include <hip/hip_bf16.h>

// ---------------------------------------------------------------------------
// RepLK-SE block for MI455X (gfx1250, wave32, WMMA).
//   x[16,256,64,64] -> dw17x17 -> pw1x1 (bf16 WMMA GEMM) -> GroupNorm(1)
//   -> GELU(exact) -> SE(avgpool->fc1->silu->fc2->sigmoid->scale) -> +x
// ---------------------------------------------------------------------------

#define Ndim 16
#define Cdim 256
#define Hdim 64
#define Wdim 64
#define HWdim (Hdim * Wdim)          // 4096
#define Kdw 17
#define PADdw 8
#define HIDden 32
#define CHW ((size_t)Cdim * HWdim)   // 1,048,576 = 2^20

typedef __bf16 v16bf __attribute__((ext_vector_type(16)));
typedef float  v8f   __attribute__((ext_vector_type(8)));
typedef int    v4i   __attribute__((ext_vector_type(4)));

union Frag16 { v16bf v; uint4 u[2]; };

// CDNA5 async global->LDS copy path (ASYNCcnt-tracked, bypasses VGPRs).
// Guarded: falls back to plain load/store if the toolchain lacks the builtins.
#if defined(__AMDGCN__) && \
    __has_builtin(__builtin_amdgcn_global_load_async_to_lds_b128) && \
    __has_builtin(__builtin_amdgcn_s_wait_asynccnt)
#define HAVE_ASYNC_LDS 1
#else
#define HAVE_ASYNC_LDS 0
#endif

#if HAVE_ASYNC_LDS
// per the hipcc diagnostic, the builtin takes int4* pointers:
//   param0: int4 AS(1)* (global src), param1: int4 AS(3)* (LDS dst)
typedef __attribute__((address_space(1))) v4i gv4i;
typedef __attribute__((address_space(3))) v4i lv4i;
#endif

__device__ __forceinline__ void stage_copy16(__bf16* dst_lds, const __bf16* src_g) {
#if HAVE_ASYNC_LDS
  __builtin_amdgcn_global_load_async_to_lds_b128(
      (gv4i*)(__UINTPTR_TYPE__)src_g, (lv4i*)(__UINTPTR_TYPE__)dst_lds, 0, 0);
#else
  *(uint4*)dst_lds = *(const uint4*)src_g;
#endif
}

// ---------------- workspace layout (bytes) ----------------
#define OFF_YDW    ((size_t)0)                       // bf16 [N][HW][C]  33,554,432
#define OFF_PWB    ((size_t)33554432)                // bf16 [C][C]         131,072
#define OFF_Y1     ((size_t)33685504)                // f32  [N][C][HW]  67,108,864
#define OFF_GSUM   ((size_t)100794368)               // f32  [N]
#define OFF_GSUMSQ ((size_t)100794624)               // f32  [N]
#define OFF_S      ((size_t)100794880)               // f32  [N][C]          16,384
#define OFF_G      ((size_t)100811264)               // f32  [N][C]          16,384

// ---------------------------------------------------------------------------
// K0: convert pointwise weights fp32 -> bf16 (tiny, 64K elems)
// ---------------------------------------------------------------------------
__global__ __launch_bounds__(256) void k0_cvt_pw(const float* __restrict__ pw,
                                                 __bf16* __restrict__ pwb) {
  int i = blockIdx.x * 256 + threadIdx.x;
  if (i < Cdim * Cdim) pwb[i] = (__bf16)pw[i];
}

// ---------------------------------------------------------------------------
// K1: depthwise 17x17, one (n,c) plane per block. 80x80 halo in LDS.
// Output channel-last bf16 [N][HW][C] so the GEMM B-tiles are contiguous
// (scattered 2B stores are merged in the 192MB L2; ydw stays L2-resident).
// ---------------------------------------------------------------------------
__global__ __launch_bounds__(256) void k1_dwconv(const float* __restrict__ x,
                                                 const float* __restrict__ dww,
                                                 __bf16* __restrict__ ydw) {
  __shared__ float sm[80 * 80];
  __shared__ float wsm[Kdw * Kdw];
  const int tid = threadIdx.x;
  const int n = blockIdx.x / Cdim;
  const int c = blockIdx.x % Cdim;
  const float* xp = x + ((size_t)n * Cdim + c) * HWdim;

  for (int i = tid; i < 80 * 80; i += 256) {
    int r = (i / 80) - PADdw;
    int q = (i % 80) - PADdw;
    float v = 0.f;
    if (r >= 0 && r < Hdim && q >= 0 && q < Wdim) v = xp[r * Wdim + q];
    sm[i] = v;
  }
  for (int i = tid; i < Kdw * Kdw; i += 256) wsm[i] = dww[c * (Kdw * Kdw) + i];
  __syncthreads();

  __bf16* op = ydw + (size_t)n * HWdim * Cdim + c;
  for (int o = tid; o < HWdim; o += 256) {
    const int oh = o >> 6, ow = o & 63;
    float acc = 0.f;
#pragma unroll
    for (int ky = 0; ky < Kdw; ++ky) {
      const float* srow = &sm[(oh + ky) * 80 + ow];
      const float* wrow = &wsm[ky * Kdw];
#pragma unroll
      for (int kx = 0; kx < Kdw; ++kx) acc = fmaf(srow[kx], wrow[kx], acc);
    }
    op[(size_t)o * Cdim] = (__bf16)acc;
  }
}

// ---------------------------------------------------------------------------
// K2: pointwise GEMM  Y1[n][co][hw] = sum_c pw[co][c] * Ydw[n][hw][c]
// bf16 WMMA 16x16x32, fp32 accumulate. Block: 8 waves, 128(co) x 64(hw) tile.
// Double-buffered LDS staging driven by GLOBAL_LOAD_ASYNC_TO_LDS_B128:
// stage s+1 streams into the other buffer while stage s feeds the WMMAs,
// synchronized with s_wait_asynccnt (6 outstanding copies per thread/stage).
// Fused: per-sample sum / sumsq for GroupNorm via atomics.
// LDS rows padded to 72 bf16 (144 B): conflict-free ds_read_b128 fragments.
// ---------------------------------------------------------------------------
#define LDP 72
__global__ __launch_bounds__(256) void k2_gemm(const __bf16* __restrict__ pwb,
                                               const __bf16* __restrict__ ydw,
                                               float* __restrict__ y1,
                                               float* __restrict__ gsum,
                                               float* __restrict__ gsumsq) {
  __shared__ __bf16 As[2][128 * LDP];   // 2 x 18.0 KB
  __shared__ __bf16 Bs[2][64 * LDP];    // 2 x  9.2 KB
  __shared__ float red[512];

  const int tid  = threadIdx.x;
  const int lane = tid & 31;
  const int wid  = tid >> 5;            // 0..7
  const int hw0  = blockIdx.x * 64;
  const int co0  = blockIdx.y * 128;
  const int n    = blockIdx.z;
  const int lr   = lane & 15;
  const int hi   = lane >> 4;           // 0 or 1

  const __bf16* bsrc = ydw + ((size_t)n * HWdim + hw0) * Cdim;

  v8f acc[4] = {};

  // issue one K-stage (A: 128x64, B: 64x64 bf16) into LDS buffer `buf`
  auto issue_stage = [&](int s, int buf) {
    const int k0 = s * 64;
#pragma unroll
    for (int e = tid * 8; e < 128 * 64; e += 256 * 8) {   // 4 copies/thread
      int row = e >> 6, kk = e & 63;
      stage_copy16(&As[buf][row * LDP + kk],
                   &pwb[(size_t)(co0 + row) * Cdim + k0 + kk]);
    }
#pragma unroll
    for (int e = tid * 8; e < 64 * 64; e += 256 * 8) {    // 2 copies/thread
      int row = e >> 6, kk = e & 63;
      stage_copy16(&Bs[buf][row * LDP + kk],
                   &bsrc[(size_t)row * Cdim + k0 + kk]);
    }
  };

  issue_stage(0, 0);

  for (int s = 0; s < 4; ++s) {
    const int cur = s & 1;
    if (s < 3) issue_stage(s + 1, cur ^ 1);
#if HAVE_ASYNC_LDS
    // oldest 6 copies (stage s) are complete; up to 6 (stage s+1) in flight
    if (s < 3) __builtin_amdgcn_s_wait_asynccnt(6);
    else       __builtin_amdgcn_s_wait_asynccnt(0);
#endif
    __syncthreads();

#pragma unroll
    for (int kk = 0; kk < 64; kk += 32) {
      // A fragment: lane<16 holds K {0..7,16..23}, lane>=16 holds {8..15,24..31}
      Frag16 af;
      const __bf16* ap = &As[cur][(wid * 16 + lr) * LDP + kk + hi * 8];
      af.u[0] = *(const uint4*)ap;
      af.u[1] = *(const uint4*)(ap + 16);
#pragma unroll
      for (int t = 0; t < 4; ++t) {
        // B fragment: lane<16 K {0..7,16..23}, lane>=16 K {8..15,24..31}, N = lane&15
        Frag16 bf;
        const __bf16* bp = &Bs[cur][(t * 16 + lr) * LDP + kk + (hi << 3)];
        bf.u[0] = *(const uint4*)bp;
        bf.u[1] = *(const uint4*)(bp + 16);
        acc[t] = __builtin_amdgcn_wmma_f32_16x16x32_bf16(
            false, af.v, false, bf.v, (short)0, acc[t], false, false);
      }
    }
    __syncthreads();   // orders reads of buf `cur` before stage s+2 overwrites it
  }

  // store D tiles + local GroupNorm partial sums
  float ls = 0.f, ls2 = 0.f;
  const int rbase = co0 + wid * 16 + hi * 8;
#pragma unroll
  for (int t = 0; t < 4; ++t) {
    const int col = hw0 + t * 16 + lr;
#pragma unroll
    for (int r = 0; r < 8; ++r) {
      float v = acc[t][r];
      y1[((size_t)n * Cdim + rbase + r) * HWdim + col] = v;
      ls += v;
      ls2 = fmaf(v, v, ls2);
    }
  }
  red[tid] = ls;
  red[256 + tid] = ls2;
  __syncthreads();
  for (int s = 128; s > 0; s >>= 1) {
    if (tid < s) {
      red[tid] += red[tid + s];
      red[256 + tid] += red[256 + tid + s];
    }
    __syncthreads();
  }
  if (tid == 0) {
    atomicAdd(&gsum[n], red[0]);
    atomicAdd(&gsumsq[n], red[256]);
  }
}

// ---------------------------------------------------------------------------
// K3: GroupNorm + exact GELU + SE global average pool. One (n,c) plane/block.
// ---------------------------------------------------------------------------
__global__ __launch_bounds__(256) void k3_pool(const float* __restrict__ y1,
                                               const float* __restrict__ gsum,
                                               const float* __restrict__ gsumsq,
                                               const float* __restrict__ gamma,
                                               const float* __restrict__ beta,
                                               float* __restrict__ sbuf) {
  __shared__ float red[256];
  const int tid = threadIdx.x;
  const int n = blockIdx.x / Cdim;
  const int c = blockIdx.x % Cdim;
  const float inv_chw = 1.0f / (float)CHW;
  const float mean = gsum[n] * inv_chw;
  const float var = gsumsq[n] * inv_chw - mean * mean;
  const float istd = rsqrtf(var + 1e-5f);
  const float ga = gamma[c], be = beta[c];

  const float* yp = y1 + ((size_t)n * Cdim + c) * HWdim;
  float ls = 0.f;
  for (int i = tid; i < HWdim; i += 256) {
    float v = fmaf((yp[i] - mean) * istd, ga, be);
    v = 0.5f * v * (1.0f + erff(v * 0.70710678118654752f));   // exact GELU
    ls += v;
  }
  red[tid] = ls;
  __syncthreads();
  for (int s = 128; s > 0; s >>= 1) {
    if (tid < s) red[tid] += red[tid + s];
    __syncthreads();
  }
  if (tid == 0) sbuf[n * Cdim + c] = red[0] * (1.0f / HWdim);
}

// ---------------------------------------------------------------------------
// K4: SE MLP (tiny): h = silu(s@fc1^T+b1); g = sigmoid(h@fc2^T+b2)
// ---------------------------------------------------------------------------
__global__ __launch_bounds__(256) void k4_mlp(const float* __restrict__ sbuf,
                                              const float* __restrict__ fc1w,
                                              const float* __restrict__ fc1b,
                                              const float* __restrict__ fc2w,
                                              const float* __restrict__ fc2b,
                                              float* __restrict__ gbuf) {
  __shared__ float hs[Ndim * HIDden];
  const int tid = threadIdx.x;
  for (int idx = tid; idx < Ndim * HIDden; idx += 256) {
    const int n = idx >> 5, j = idx & 31;
    float a = fc1b[j];
    const float* sp = &sbuf[n * Cdim];
    const float* wp = &fc1w[j * Cdim];
    for (int c = 0; c < Cdim; ++c) a = fmaf(sp[c], wp[c], a);
    hs[idx] = a / (1.0f + expf(-a));            // silu
  }
  __syncthreads();
  for (int idx = tid; idx < Ndim * Cdim; idx += 256) {
    const int n = idx >> 8, c = idx & 255;
    float a = fc2b[c];
    const float* hp = &hs[n * HIDden];
    const float* wp = &fc2w[c * HIDden];
#pragma unroll
    for (int j = 0; j < HIDden; ++j) a = fmaf(hp[j], wp[j], a);
    gbuf[idx] = 1.0f / (1.0f + expf(-a));       // sigmoid
  }
}

// ---------------------------------------------------------------------------
// K5: out = x + gelu(norm(y1)) * g[n][c]   (float4 vectorized; recompute
// norm+gelu instead of re-storing a 67MB tensor -> saves 134MB of traffic)
// ---------------------------------------------------------------------------
__global__ __launch_bounds__(256) void k5_final(const float* __restrict__ x,
                                                const float* __restrict__ y1,
                                                const float* __restrict__ gsum,
                                                const float* __restrict__ gsumsq,
                                                const float* __restrict__ gamma,
                                                const float* __restrict__ beta,
                                                const float* __restrict__ gbuf,
                                                float* __restrict__ out) {
  const size_t idx = (size_t)blockIdx.x * 256 + threadIdx.x;   // float4 index
  const size_t flat = idx * 4;
  const int n = (int)(flat >> 20);            // C*HW = 2^20
  const int c = (int)((flat >> 12) & 255);    // HW   = 2^12
  const float inv_chw = 1.0f / (float)CHW;
  const float mean = gsum[n] * inv_chw;
  const float var = gsumsq[n] * inv_chw - mean * mean;
  const float istd = rsqrtf(var + 1e-5f);
  const float ga = gamma[c], be = beta[c];
  const float sc = gbuf[n * Cdim + c];

  const float4 xv = ((const float4*)x)[idx];
  const float4 yv = ((const float4*)y1)[idx];
  float4 ov;
#define GELU_SCALE(Y, X)                                             \
  {                                                                  \
    float t = fmaf(((Y) - mean) * istd, ga, be);                     \
    t = 0.5f * t * (1.0f + erff(t * 0.70710678118654752f));          \
    t = fmaf(t, sc, (X));                                            \
    (Y) = t;                                                         \
  }
  ov = yv;
  GELU_SCALE(ov.x, xv.x);
  GELU_SCALE(ov.y, xv.y);
  GELU_SCALE(ov.z, xv.z);
  GELU_SCALE(ov.w, xv.w);
#undef GELU_SCALE
  ((float4*)out)[idx] = ov;
}

// ---------------------------------------------------------------------------
extern "C" void kernel_launch(void* const* d_in, const int* in_sizes, int n_in,
                              void* d_out, int out_size, void* d_ws, size_t ws_size,
                              hipStream_t stream) {
  const float* x     = (const float*)d_in[0];
  const float* dw_w  = (const float*)d_in[1];
  const float* pw_w  = (const float*)d_in[2];
  const float* gamma = (const float*)d_in[3];
  const float* beta  = (const float*)d_in[4];
  const float* fc1w  = (const float*)d_in[5];
  const float* fc1b  = (const float*)d_in[6];
  const float* fc2w  = (const float*)d_in[7];
  const float* fc2b  = (const float*)d_in[8];
  float* out = (float*)d_out;

  char* ws = (char*)d_ws;
  __bf16* ydw   = (__bf16*)(ws + OFF_YDW);
  __bf16* pwb   = (__bf16*)(ws + OFF_PWB);
  float* y1     = (float*)(ws + OFF_Y1);
  float* gsum   = (float*)(ws + OFF_GSUM);
  float* gsumsq = (float*)(ws + OFF_GSUMSQ);
  float* sbuf   = (float*)(ws + OFF_S);
  float* gbuf   = (float*)(ws + OFF_G);

  // zero GroupNorm accumulators (graph-capture safe async memset)
  (void)hipMemsetAsync(ws + OFF_GSUM, 0, 512, stream);

  k0_cvt_pw<<<(Cdim * Cdim + 255) / 256, 256, 0, stream>>>(pw_w, pwb);
  k1_dwconv<<<Ndim * Cdim, 256, 0, stream>>>(x, dw_w, ydw);
  {
    dim3 grid(HWdim / 64, Cdim / 128, Ndim);   // 64 x 2 x 16
    k2_gemm<<<grid, 256, 0, stream>>>(pwb, ydw, y1, gsum, gsumsq);
  }
  k3_pool<<<Ndim * Cdim, 256, 0, stream>>>(y1, gsum, gsumsq, gamma, beta, sbuf);
  k4_mlp<<<1, 256, 0, stream>>>(sbuf, fc1w, fc1b, fc2w, fc2b, gbuf);
  {
    const size_t n4 = ((size_t)Ndim * CHW) / 4;          // 4,194,304
    k5_final<<<(unsigned)(n4 / 256), 256, 0, stream>>>(x, y1, gsum, gsumsq,
                                                       gamma, beta, gbuf, out);
  }
}
